// MultiHeadAttention_26259430048158
// MI455X (gfx1250) — matrix operand order repaired and checked by reading, not compiled, not run on hardware
//
#include <hip/hip_runtime.h>
#include <hip/hip_bf16.h>

// Graph-transformer layer for MI455X (gfx1250, wave32).
// GEMMs: bf16 WMMA pipe (v_wmma_f32_16x16x32_bf16, f32 accum), 16x64 tile per
// wave (one A fragment feeds 4 WMMAs). Edge phase: wave-per-edge, L2-resident
// gathers, scalarized index loads, native f32 atomics.

typedef __attribute__((ext_vector_type(8)))  __bf16 v8bf;
typedef __attribute__((ext_vector_type(16))) __bf16 v16bf;
typedef __attribute__((ext_vector_type(8)))  float  v8f;

#define DMODEL 128
#define NHEAD  8
#define DHEAD  16
#define DFF    512
#define CLAMPV 5.0f

// ---------------------------------------------------------------- convert
__global__ void k_f2b(const float* __restrict__ in, __bf16* __restrict__ out, int n) {
  int i = blockIdx.x * blockDim.x + threadIdx.x;
  if (i < n) out[i] = (__bf16)in[i];
}

__global__ void k_zero(float* __restrict__ p, int n) {
  int i = blockIdx.x * blockDim.x + threadIdx.x;
  if (i < n) p[i] = 0.0f;
}

// ---------------------------------------------------------------- WMMA GEMM
// C[M,NOUT] = A[M,K] (bf16 row-major) @ W[NOUT,K]^T (bf16 row-major weight).
// Per-wave tile: 16 rows x 64 cols (4 accumulators share one A fragment).
// Block: 4 waves stacked along M -> 64x64 block tile (waves share B addresses).
// A-frag: lane<16 holds row m0+lane, K = k0+0..15 contiguous;
//         lane>=16 holds row m0+(lane-16), K = k0+16..31 contiguous.
// B-frag: lane<16 holds col n+lane (= weight row), K = k0+0..15 contiguous;
//         lane>=16 same col, K = k0+16..31 contiguous. (32B contiguous loads)
template <bool HAS_BIAS, bool RELU, bool OUT_BF16>
__global__ void k_gemm_wmma(const __bf16* __restrict__ A,
                            const __bf16* __restrict__ W,
                            const float* __restrict__ bias,
                            void* __restrict__ out,
                            int K, int NOUT) {
  const int lane = threadIdx.x & 31;
  const int wave = threadIdx.x >> 5;
  const int m0   = (blockIdx.x << 2 | wave) << 4;  // 64-row block tile
  const int n0   = blockIdx.y << 6;                // 64-col block tile
  const int lo16 = lane & 15;
  const int hi   = lane >> 4;                      // 0 or 1

  const __bf16* pA  = A + (size_t)(m0 + lo16) * K + hi * 16;
  const __bf16* pB0 = W + (size_t)(n0 + lo16) * K + hi * 16;
  const __bf16* pB1 = pB0 + (size_t)16 * K;
  const __bf16* pB2 = pB0 + (size_t)32 * K;
  const __bf16* pB3 = pB0 + (size_t)48 * K;

  v8f c0 = {}, c1 = {}, c2 = {}, c3 = {};
  for (int k0 = 0; k0 < K; k0 += 32) {
    v8bf alo = *(const v8bf*)(pA + k0);
    v8bf ahi = *(const v8bf*)(pA + k0 + 8);
    v16bf a = __builtin_shufflevector(alo, ahi, 0,1,2,3,4,5,6,7,8,9,10,11,12,13,14,15);
    v16bf b0 = *(const v16bf*)(pB0 + k0);
    v16bf b1 = *(const v16bf*)(pB1 + k0);
    v16bf b2 = *(const v16bf*)(pB2 + k0);
    v16bf b3 = *(const v16bf*)(pB3 + k0);
    c0 = __builtin_amdgcn_wmma_f32_16x16x32_bf16(false, a, false, b0, (short)0, c0, false, false);
    c1 = __builtin_amdgcn_wmma_f32_16x16x32_bf16(false, a, false, b1, (short)0, c1, false, false);
    c2 = __builtin_amdgcn_wmma_f32_16x16x32_bf16(false, a, false, b2, (short)0, c2, false, false);
    c3 = __builtin_amdgcn_wmma_f32_16x16x32_bf16(false, a, false, b3, (short)0, c3, false, false);
  }

  // C/D layout: VGPR r, lanes 0-15 -> (M=m0+r, N=n+lane); lanes 16-31 -> M=m0+8+r.
  const int rbase = m0 + hi * 8;
  v8f acc[4] = {c0, c1, c2, c3};
#pragma unroll
  for (int t = 0; t < 4; ++t) {
    const int col = n0 + t * 16 + lo16;
    float bs = HAS_BIAS ? bias[col] : 0.0f;
#pragma unroll
    for (int r = 0; r < 8; ++r) {
      float v = acc[t][r] + bs;
      if (RELU) v = fmaxf(v, 0.0f);
      size_t idx = (size_t)(rbase + r) * NOUT + col;
      if (OUT_BF16) ((__bf16*)out)[idx] = (__bf16)v;
      else          ((float*)out)[idx]  = v;
    }
  }
}

// ---------------------------------------------------------------- edge phase
// qkv layout per node: [0..127]=q, [128..255]=k, [256..383]=v  (f32)
// Logits are clamped to +-CLAMPV, so softmax uses the constant shift
// exp(u - CLAMPV) (shift-invariant, numerically safe) -> no segment_max pass.
// One wave per edge: e is wave-uniform, so src/dst loads scalarize; each lane
// handles 4 contiguous components; head-dot reduces within 4-lane groups.
__global__ void k_edge_logits(const float* __restrict__ qkv,
                              const int* __restrict__ src,
                              const int* __restrict__ dst,
                              float* __restrict__ wbuf,     // [E,H]
                              float* __restrict__ denom,    // [N,H]
                              int E_) {
  const int lane = threadIdx.x & 31;
  const int e = blockIdx.x * (blockDim.x >> 5) + (threadIdx.x >> 5);
  if (e >= E_) return;
  const int s = src[e], d = dst[e];            // wave-uniform
  float4 q = ((const float4*)(qkv + (size_t)d * 384))[lane];
  float4 k = ((const float4*)(qkv + (size_t)s * 384 + DMODEL))[lane];
  float p = q.x * k.x + q.y * k.y + q.z * k.z + q.w * k.w;
  // head h = lane>>2 spans lanes 4h..4h+3 (16 components)
  p += __shfl_xor(p, 1, 32);
  p += __shfl_xor(p, 2, 32);
  if ((lane & 3) == 0) {
    float u = p * 0.25f;                       // 1/sqrt(DHEAD)
    u = fminf(fmaxf(u, -CLAMPV), CLAMPV);
    float w = __expf(u - CLAMPV);
    int h = lane >> 2;
    wbuf[(size_t)e * NHEAD + h] = w;
    unsafeAtomicAdd(&denom[d * NHEAD + h], w);
  }
}

__global__ void k_edge_scatter(const float* __restrict__ qkv,
                               const int* __restrict__ src,
                               const int* __restrict__ dst,
                               const float* __restrict__ wbuf,
                               const float* __restrict__ denom,
                               float* __restrict__ sumav,   // [N,128]
                               int E_) {
  const int lane = threadIdx.x & 31;
  const int e = blockIdx.x * (blockDim.x >> 5) + (threadIdx.x >> 5);
  if (e >= E_) return;
  const int s = src[e], d = dst[e];            // wave-uniform
  const int h = lane >> 2;                     // 4 lanes per head
  float a = wbuf[(size_t)e * NHEAD + h] / denom[d * NHEAD + h];
  float4 v = ((const float4*)(qkv + (size_t)s * 384 + 256))[lane];
  float* o = sumav + (size_t)d * DMODEL + lane * 4;
  unsafeAtomicAdd(o + 0, a * v.x);
  unsafeAtomicAdd(o + 1, a * v.y);
  unsafeAtomicAdd(o + 2, a * v.z);
  unsafeAtomicAdd(o + 3, a * v.w);
}

// ---------------------------------------------------------------- layernorm
// One wave per node, 4 components per lane, wave32 shuffle reductions.
__global__ void k_ln_residual(const float* __restrict__ x1,
                              const float* __restrict__ x2,
                              const float* __restrict__ g,
                              const float* __restrict__ b,
                              float* __restrict__ of,       // f32 out (or null)
                              __bf16* __restrict__ ob,      // bf16 out (or null)
                              int nn) {
  const int lane = threadIdx.x & 31;
  const int node = blockIdx.x * (blockDim.x >> 5) + (threadIdx.x >> 5);
  if (node >= nn) return;
  float4 a = ((const float4*)(x1 + (size_t)node * DMODEL))[lane];
  float4 c = ((const float4*)(x2 + (size_t)node * DMODEL))[lane];
  float v0 = a.x + c.x, v1 = a.y + c.y, v2 = a.z + c.z, v3 = a.w + c.w;
  float s = v0 + v1 + v2 + v3;
#pragma unroll
  for (int off = 16; off >= 1; off >>= 1) s += __shfl_xor(s, off, 32);
  float mu = s * (1.0f / 128.0f);
  float d0 = v0 - mu, d1 = v1 - mu, d2 = v2 - mu, d3 = v3 - mu;
  float q = d0 * d0 + d1 * d1 + d2 * d2 + d3 * d3;
#pragma unroll
  for (int off = 16; off >= 1; off >>= 1) q += __shfl_xor(q, off, 32);
  float inv = rsqrtf(q * (1.0f / 128.0f) + 1e-5f);
  float4 gg = ((const float4*)g)[lane];
  float4 bb = ((const float4*)b)[lane];
  float4 y;
  y.x = d0 * inv * gg.x + bb.x;
  y.y = d1 * inv * gg.y + bb.y;
  y.z = d2 * inv * gg.z + bb.z;
  y.w = d3 * inv * gg.w + bb.w;
  if (of) ((float4*)(of + (size_t)node * DMODEL))[lane] = y;
  if (ob) {
    __bf16* p = ob + (size_t)node * DMODEL + lane * 4;
    p[0] = (__bf16)y.x; p[1] = (__bf16)y.y; p[2] = (__bf16)y.z; p[3] = (__bf16)y.w;
  }
}

// ---------------------------------------------------------------- launch
static inline int ceil_div(int a, int b) { return (a + b - 1) / b; }

extern "C" void kernel_launch(void* const* d_in, const int* in_sizes, int n_in,
                              void* d_out, int out_size, void* d_ws, size_t ws_size,
                              hipStream_t stream) {
  (void)n_in; (void)out_size; (void)ws_size;
  const float* feat = (const float*)d_in[0];
  const int*   src  = (const int*)d_in[1];
  const int*   dst  = (const int*)d_in[2];
  const float* Wq   = (const float*)d_in[3];
  const float* Wk   = (const float*)d_in[4];
  const float* Wv   = (const float*)d_in[5];
  const float* Wo   = (const float*)d_in[6];
  const float* ln1g = (const float*)d_in[7];
  const float* ln1b = (const float*)d_in[8];
  const float* W1   = (const float*)d_in[9];
  const float* b1   = (const float*)d_in[10];
  const float* W2   = (const float*)d_in[11];
  const float* b2   = (const float*)d_in[12];
  const float* ln2g = (const float*)d_in[13];
  const float* ln2b = (const float*)d_in[14];

  const int N = in_sizes[0] / DMODEL;   // 40000 (multiple of 64)
  const int E = in_sizes[1];            // 640000

  // ---- workspace arena (256B aligned slots, with lifetime-based reuse) ----
  char* ws = (char*)d_ws;
  size_t off = 0;
  auto alloc = [&](size_t bytes) {
    size_t o = off; off += (bytes + 255) & ~(size_t)255; return o;
  };
  __bf16* featb  = (__bf16*)(ws + alloc((size_t)N * DMODEL * 2));
  __bf16* wqkvb  = (__bf16*)(ws + alloc((size_t)384 * DMODEL * 2));
  __bf16* wob    = (__bf16*)(ws + alloc((size_t)DMODEL * DMODEL * 2));
  __bf16* w1b    = (__bf16*)(ws + alloc((size_t)DFF * DMODEL * 2));
  __bf16* w2b    = (__bf16*)(ws + alloc((size_t)DMODEL * DFF * 2));
  size_t  o_qkv  = alloc((size_t)N * 384 * 4);        // reused: uh + midb
  size_t  o_wbuf = alloc((size_t)E * NHEAD * 4);      // reused: ffn_out
  float*  denom  = (float*)(ws + alloc((size_t)N * NHEAD * 4));
  size_t  o_sav  = alloc((size_t)N * DMODEL * 4);     // reused: h1 (f32)
  size_t  o_savb = alloc((size_t)N * DMODEL * 2);     // reused: h1 (bf16)

  float*  qkv    = (float*)(ws + o_qkv);
  float*  wbuf   = (float*)(ws + o_wbuf);
  float*  sumav  = (float*)(ws + o_sav);
  __bf16* sumavb = (__bf16*)(ws + o_savb);
  // reuse (non-overlapping lifetimes):
  float*  uh     = (float*)(ws + o_qkv);                            // after edge phase
  __bf16* midb   = (__bf16*)(ws + o_qkv + (size_t)N * DMODEL * 4);  // FFN hidden
  float*  ffnout = (float*)(ws + o_wbuf);                           // after edge phase
  float*  h1     = (float*)(ws + o_sav);                            // after sumav->bf16
  __bf16* h1b    = (__bf16*)(ws + o_savb);                          // after Wo GEMM

  const int Mt64 = N / 64;  // 625 M-supertiles (64 rows each)

  // 1) convert activations + weights to bf16
  {
    int n = N * DMODEL;
    k_f2b<<<ceil_div(n, 256), 256, 0, stream>>>(feat, featb, n);
    k_f2b<<<ceil_div(DMODEL * DMODEL, 256), 256, 0, stream>>>(Wq, wqkvb, DMODEL * DMODEL);
    k_f2b<<<ceil_div(DMODEL * DMODEL, 256), 256, 0, stream>>>(Wk, wqkvb + DMODEL * DMODEL, DMODEL * DMODEL);
    k_f2b<<<ceil_div(DMODEL * DMODEL, 256), 256, 0, stream>>>(Wv, wqkvb + 2 * DMODEL * DMODEL, DMODEL * DMODEL);
    k_f2b<<<ceil_div(DMODEL * DMODEL, 256), 256, 0, stream>>>(Wo, wob, DMODEL * DMODEL);
    k_f2b<<<ceil_div(DFF * DMODEL, 256), 256, 0, stream>>>(W1, w1b, DFF * DMODEL);
    k_f2b<<<ceil_div(DMODEL * DFF, 256), 256, 0, stream>>>(W2, w2b, DMODEL * DFF);
  }

  // 2) fused QKV projection: [N,128] @ [384,128]^T -> qkv [N,384]
  k_gemm_wmma<false, false, false><<<dim3(Mt64, 384 / 64), 128, 0, stream>>>(
      featb, wqkvb, nullptr, qkv, DMODEL, 384);

  // 3) zero accumulators (d_ws is poisoned; must re-zero every call)
  k_zero<<<ceil_div(N * NHEAD, 256), 256, 0, stream>>>(denom, N * NHEAD);
  k_zero<<<ceil_div(N * DMODEL, 256), 256, 0, stream>>>(sumav, N * DMODEL);

  // 4) edge logits + softmax denominators (wave per edge, L2-resident gathers)
  k_edge_logits<<<ceil_div(E, 8), 256, 0, stream>>>(qkv, src, dst, wbuf, denom, E);

  // 5) weighted-V scatter-reduce (wave per edge)
  k_edge_scatter<<<ceil_div(E, 8), 256, 0, stream>>>(qkv, src, dst, wbuf, denom, sumav, E);

  // 6) output projection: sumav -> bf16, then [N,128] @ Wo^T -> uh
  k_f2b<<<ceil_div(N * DMODEL, 256), 256, 0, stream>>>(sumav, sumavb, N * DMODEL);
  k_gemm_wmma<false, false, false><<<dim3(Mt64, DMODEL / 64), 128, 0, stream>>>(
      sumavb, wob, nullptr, uh, DMODEL, DMODEL);

  // 7) LN1(feat + uh) -> h1 (f32 + bf16)
  k_ln_residual<<<ceil_div(N, 8), 256, 0, stream>>>(feat, uh, ln1g, ln1b, h1, h1b, N);

  // 8) FFN: relu(h1 @ W1^T + b1) @ W2^T + b2
  k_gemm_wmma<true, true, true><<<dim3(Mt64, DFF / 64), 128, 0, stream>>>(
      h1b, w1b, b1, midb, DMODEL, DFF);
  k_gemm_wmma<true, false, false><<<dim3(Mt64, DMODEL / 64), 128, 0, stream>>>(
      midb, w2b, b2, ffnout, DFF, DMODEL);

  // 9) LN2(h1 + ffn) -> out
  k_ln_residual<<<ceil_div(N, 8), 256, 0, stream>>>(h1, ffnout, ln2g, ln2b,
                                                    (float*)d_out, nullptr, N);
}
